// GNN_cell_33019708572246
// MI455X (gfx1250) — matrix operand order repaired and checked by reading, not compiled, not run on hardware
//
#include <hip/hip_runtime.h>
#include <math.h>

// ---------------------------------------------------------------------------
// GNN cell (3x GATConv + max_pool + BatchNorm) for MI455X / gfx1250.
// Dense GEMMs run on the WMMA pipe (v_wmma_f32_16x16x32_bf16, f32 accum),
// with LDS tiles stored in fragment order so each lane's operands are
// contiguous (ds_load_b128 instead of scalar ds_load_u16 gathers).
// Scatter softmax uses hardware atomics (u32 ordered-max, f32 add).
// ---------------------------------------------------------------------------

typedef __attribute__((ext_vector_type(16))) __bf16 v16bf;
typedef __attribute__((ext_vector_type(8)))  __bf16 v8bf;
typedef __attribute__((ext_vector_type(8)))  float  v8f;

#define DIM   128
#define NEDGE 262144
#define HASH_CAP (1u << 20)
#define EPSV 1e-5f

__device__ __forceinline__ float leaky02(float v) { return v > 0.f ? v : 0.2f * v; }
// Order-preserving float <-> u32 encoding so we can use global_atomic_max_u32.
__device__ __forceinline__ unsigned encf(float f) {
    unsigned u = __float_as_uint(f);
    return (u & 0x80000000u) ? ~u : (u | 0x80000000u);
}
__device__ __forceinline__ float decf(unsigned e) {
    return __uint_as_float((e & 0x80000000u) ? (e & 0x7fffffffu) : ~e);
}

// f32 -> bf16 conversion (done once per layer for X and W).
__global__ void to_bf16(const float* __restrict__ in, __bf16* __restrict__ out, int n) {
    int i = blockIdx.x * blockDim.x + threadIdx.x;
    if (i < n) out[i] = (__bf16)in[i];
}

// ---------------------------------------------------------------------------
// H[M x 128] = Xb[M x K] * Wb[K x 128], bf16 in, f32 accumulate via WMMA.
// Block = 256 threads (8 waves). blockIdx.x = M-tile (16 rows).
// Wave w computes the 16x16 output tile at columns [16w, 16w+16).
// K is a template constant (64 or 128) so the K-loop fully unrolls.
// ---------------------------------------------------------------------------
template <int K>
__global__ __launch_bounds__(256)
void gemm_wmma_bf16(const __bf16* __restrict__ Xb, const __bf16* __restrict__ Wb,
                    float* __restrict__ H, int M) {
    __shared__ __attribute__((aligned(32))) __bf16 As[16 * 32];    // [m][k] row-major
    __shared__ __attribute__((aligned(32))) __bf16 Bst[128 * 32];  // [n][k] (transposed)
    const int tid  = threadIdx.x;
    const int lane = tid & 31;
    const int wv   = tid >> 5;        // 0..7 -> N tile
    const int m0   = blockIdx.x * 16;
    const int half = lane >> 4;       // 0 or 1
    const int l16  = lane & 15;

    v8f c = {0.f, 0.f, 0.f, 0.f, 0.f, 0.f, 0.f, 0.f};

#pragma unroll
    for (int kc = 0; kc < K; kc += 32) {
        // Stage A tile (16x32 bf16 = 1KB) as 256 dword copies, coalesced.
        {
            int r  = tid >> 4;             // 0..15
            int c2 = (tid & 15) * 2;       // 0,2,..,30
            *(unsigned*)&As[r * 32 + c2] =
                *(const unsigned*)&Xb[(size_t)(m0 + r) * K + (kc + c2)];
        }
        // Stage B tile (32x128 bf16 = 8KB) transposed to [n][k].
        for (int t = tid; t < 4096; t += 256) {
            int kk = t >> 7, nn = t & 127;                 // coalesced global reads
            Bst[nn * 32 + kk] = Wb[(size_t)(kc + kk) * 128 + nn];
        }
        if (kc + 32 < K)  // prefetch next X chunk (global_prefetch_b8)
            __builtin_prefetch(&Xb[(size_t)(m0 + l16) * K + kc + 32], 0, 3);
        __syncthreads();

        // A fragment: lane(half,l16) uses K = half*16 .. half*16+15
        // (same k order as B): one contiguous 32B run.
        v16bf a = *(const v16bf*)&As[l16 * 32 + half * 16];
        // B fragment: lane(half,n=l16) needs K = half*16 .. half*16+15:
        // one contiguous 32B run in the transposed tile.
        v16bf b = *(const v16bf*)&Bst[(wv * 16 + l16) * 32 + half * 16];

        c = __builtin_amdgcn_wmma_f32_16x16x32_bf16(false, a, false, b,
                                                    (short)0, c, false, false);
        __syncthreads();
    }
    // D 16x16 f32 layout: VGPR r -> M = r + 8*half, N = l16.
#pragma unroll
    for (int r = 0; r < 8; ++r) {
        int m = m0 + r + half * 8;
        int n = wv * 16 + l16;
        H[(size_t)m * 128 + n] = c[r];
    }
}

// es[i] = h[i].a_src ; ed[i] = h[i].a_dst
__global__ void attn_terms(const float* __restrict__ H, const float* __restrict__ as,
                           const float* __restrict__ ad, float* es, float* ed, int M) {
    int i = blockIdx.x * blockDim.x + threadIdx.x;
    if (i >= M) return;
    const float* h = H + (size_t)i * DIM;
    float s = 0.f, d = 0.f;
#pragma unroll 8
    for (int f = 0; f < DIM; ++f) { float v = h[f]; s += v * as[f]; d += v * ad[f]; }
    es[i] = s; ed[i] = d;
}

__global__ void init_mask(const int* __restrict__ src, const int* __restrict__ dst,
                          int* mask, int E) {
    int e = blockIdx.x * blockDim.x + threadIdx.x;
    if (e < E) mask[e] = (src[e] != dst[e]) ? 1 : 0;
}

// Self-loop logit; seeds the running max.
__global__ void init_self(const float* es, const float* ed, float* selfl,
                          unsigned* mxE, int M) {
    int i = blockIdx.x * blockDim.x + threadIdx.x;
    if (i >= M) return;
    float l = leaky02(es[i] + ed[i]);
    selfl[i] = l;
    mxE[i] = encf(l);
}

__global__ void edge_max(const int* __restrict__ src, const int* __restrict__ dst,
                         const int* __restrict__ mask, const float* __restrict__ es,
                         const float* __restrict__ ed, unsigned* mxE, int E) {
    int e = blockIdx.x * blockDim.x + threadIdx.x;
    if (e >= E || !mask[e]) return;
    float l = leaky02(es[src[e]] + ed[dst[e]]);
    atomicMax(&mxE[dst[e]], encf(l));
}

// den/num seeded with self-loop contribution.
__global__ void init_accum(const float* __restrict__ selfl, const unsigned* __restrict__ mxE,
                           const float* __restrict__ H, float* den, float* num, int M) {
    int idx = blockIdx.x * blockDim.x + threadIdx.x;
    if (idx >= M * DIM) return;
    int i = idx >> 7, f = idx & 127;
    float ex = __expf(selfl[i] - decf(mxE[i]));
    num[idx] = ex * H[(size_t)i * DIM + f];
    if (f == 0) den[i] = ex;
}

// One thread per (edge, 4-feature group): 32 threads per edge.
__global__ void edge_accum(const int* __restrict__ src, const int* __restrict__ dst,
                           const int* __restrict__ mask, const float* __restrict__ es,
                           const float* __restrict__ ed, const unsigned* __restrict__ mxE,
                           const float* __restrict__ H, float* den, float* num, int E) {
    int idx = blockIdx.x * blockDim.x + threadIdx.x;
    int e = idx >> 5;
    if (e >= E || !mask[e]) return;
    int s = src[e], d = dst[e];
    float ex = __expf(leaky02(es[s] + ed[d]) - decf(mxE[d]));
    int fo = (idx & 31) << 2;
    if (fo == 0) atomicAdd(&den[d], ex);
    const float* hs = H + (size_t)s * DIM;
    float* nd = num + (size_t)d * DIM;
#pragma unroll
    for (int j = 0; j < 4; ++j) atomicAdd(&nd[fo + j], ex * hs[fo + j]);
}

__global__ void finalize_relu(float* num, const float* __restrict__ den,
                              const float* __restrict__ bias, int M) {
    int idx = blockIdx.x * blockDim.x + threadIdx.x;
    if (idx >= M * DIM) return;
    int i = idx >> 7, f = idx & 127;
    float v = num[idx] / den[i] + bias[f];
    num[idx] = v > 0.f ? v : 0.f;
}

// Cluster maps are arange//2 with per-graph offsets => globally node -> node>>1.
__global__ void pool2(const float* __restrict__ xin, float* xout, int Mout) {
    int idx = blockIdx.x * blockDim.x + threadIdx.x;
    if (idx >= Mout * DIM) return;
    int i = idx >> 7, f = idx & 127;
    float a = xin[(size_t)(2 * i) * DIM + f];
    float b = xin[(size_t)(2 * i + 1) * DIM + f];
    xout[idx] = fmaxf(a, b);
}

__global__ void hash_clear(unsigned* tab, int cap) {
    int i = blockIdx.x * blockDim.x + threadIdx.x;
    if (i < cap) tab[i] = 0xffffffffu;
}

// remap edges through pooling (>>1), drop self loops + duplicates via hash set.
__global__ void remap_dedupe(const int* __restrict__ src, const int* __restrict__ dst,
                             const int* __restrict__ mask, int* nsrc, int* ndst,
                             int* nmask, unsigned* tab, unsigned Mnew, int E) {
    int e = blockIdx.x * blockDim.x + threadIdx.x;
    if (e >= E) return;
    int s = src[e] >> 1, d = dst[e] >> 1;
    nsrc[e] = s; ndst[e] = d;
    int m = mask[e] && (s != d);
    if (m) {
        unsigned key = (unsigned)s * Mnew + (unsigned)d;   // < 2^28, never 0xFFFFFFFF
        unsigned p = (key * 2654435761u) & (HASH_CAP - 1u);
        for (;;) {
            unsigned old = atomicCAS(&tab[p], 0xffffffffu, key);
            if (old == 0xffffffffu) break;          // inserted: keep this edge
            if (old == key) { m = 0; break; }        // duplicate: drop
            p = (p + 1) & (HASH_CAP - 1u);
        }
    }
    nmask[e] = m;
}

__global__ void bn_stats(const float* __restrict__ X, float* mu, float* var, int M) {
    __shared__ float ss[256], sq[256];
    int f = blockIdx.x, t = threadIdx.x;
    float s = 0.f, q = 0.f;
    for (int i = t; i < M; i += 256) {
        float v = X[(size_t)i * DIM + f];
        s += v; q += v * v;
    }
    ss[t] = s; sq[t] = q;
    __syncthreads();
    for (int o = 128; o > 0; o >>= 1) {
        if (t < o) { ss[t] += ss[t + o]; sq[t] += sq[t + o]; }
        __syncthreads();
    }
    if (t == 0) {
        float m = ss[0] / (float)M;
        mu[f] = m;
        var[f] = sq[0] / (float)M - m * m;
    }
}

__global__ void bn_apply(float* X, const float* __restrict__ mu,
                         const float* __restrict__ var, int M) {
    int idx = blockIdx.x * blockDim.x + threadIdx.x;
    if (idx >= M * DIM) return;
    int f = idx & 127;
    X[idx] = (X[idx] - mu[f]) * rsqrtf(var[f] + EPSV);
}

// ---------------------------------------------------------------------------
extern "C" void kernel_launch(void* const* d_in, const int* in_sizes, int n_in,
                              void* d_out, int out_size, void* d_ws, size_t ws_size,
                              hipStream_t stream) {
    const float* x0   = (const float*)d_in[0];   // [32768, 64]
    const float* Ws[3] = { (const float*)d_in[1], (const float*)d_in[2], (const float*)d_in[3] };
    const float* atts = (const float*)d_in[4];   // [3,128]
    const float* attd = (const float*)d_in[5];   // [3,128]
    const float* bias = (const float*)d_in[6];   // [3,128]
    const int*   src0 = (const int*)d_in[7];     // edge_index row 0
    const int*   dst0 = src0 + NEDGE;            // edge_index row 1

    // Workspace carve-up (~61 MB total).
    char* wsb = (char*)d_ws;
    size_t o = 0;
    float* h    = (float*)(wsb + o); o += (size_t)32768 * DIM * 4;   // 16 MB
    float* num  = (float*)(wsb + o); o += (size_t)32768 * DIM * 4;   // 16 MB
    float* xp   = (float*)(wsb + o); o += (size_t)16384 * DIM * 4;   //  8 MB
    __bf16* Xb  = (__bf16*)(wsb + o); o += (size_t)32768 * DIM * 2;  //  8 MB
    __bf16* Wb  = (__bf16*)(wsb + o); o += (size_t)DIM * DIM * 2;    // 32 KB
    float* es   = (float*)(wsb + o); o += 32768 * 4;
    float* ed   = (float*)(wsb + o); o += 32768 * 4;
    float* sfl  = (float*)(wsb + o); o += 32768 * 4;
    unsigned* mxE = (unsigned*)(wsb + o); o += 32768 * 4;
    float* den  = (float*)(wsb + o); o += 32768 * 4;
    int* srcA   = (int*)(wsb + o); o += NEDGE * 4;
    int* dstA   = (int*)(wsb + o); o += NEDGE * 4;
    int* mskA   = (int*)(wsb + o); o += NEDGE * 4;
    int* srcB   = (int*)(wsb + o); o += NEDGE * 4;
    int* dstB   = (int*)(wsb + o); o += NEDGE * 4;
    int* mskB   = (int*)(wsb + o); o += NEDGE * 4;
    unsigned* tab = (unsigned*)(wsb + o); o += (size_t)HASH_CAP * 4; //  4 MB
    float* mu   = (float*)(wsb + o); o += DIM * 4;
    float* var  = (float*)(wsb + o); o += DIM * 4;
    (void)ws_size; (void)in_sizes; (void)n_in; (void)out_size;

    const int E = NEDGE;
    init_mask<<<(E + 255) / 256, 256, 0, stream>>>(src0, dst0, mskA, E);

    const int* src = src0;
    const int* dst = dst0;
    const int* msk = mskA;
    const float* x = x0;
    int M = 32768, K = 64;

    for (int layer = 0; layer < 3; ++layer) {
        // bf16 staging buffers for the WMMA GEMM.
        to_bf16<<<(M * K + 255) / 256, 256, 0, stream>>>(x, Xb, M * K);
        to_bf16<<<(K * DIM + 255) / 256, 256, 0, stream>>>(Ws[layer], Wb, K * DIM);
        if (K == 64)
            gemm_wmma_bf16<64><<<M / 16, 256, 0, stream>>>(Xb, Wb, h, M);
        else
            gemm_wmma_bf16<128><<<M / 16, 256, 0, stream>>>(Xb, Wb, h, M);

        attn_terms<<<(M + 255) / 256, 256, 0, stream>>>(h, atts + layer * DIM,
                                                        attd + layer * DIM, es, ed, M);
        init_self<<<(M + 255) / 256, 256, 0, stream>>>(es, ed, sfl, mxE, M);
        edge_max<<<(E + 255) / 256, 256, 0, stream>>>(src, dst, msk, es, ed, mxE, E);
        init_accum<<<(M * DIM + 255) / 256, 256, 0, stream>>>(sfl, mxE, h, den, num, M);
        edge_accum<<<(E * 32 + 255) / 256, 256, 0, stream>>>(src, dst, msk, es, ed,
                                                             mxE, h, den, num, E);
        finalize_relu<<<(M * DIM + 255) / 256, 256, 0, stream>>>(num, den,
                                                                 bias + layer * DIM, M);
        int Mn = M / 2;
        float* pooled = (layer == 2) ? (float*)d_out : xp;
        pool2<<<(Mn * DIM + 255) / 256, 256, 0, stream>>>(num, pooled, Mn);
        bn_stats<<<DIM, 256, 0, stream>>>(pooled, mu, var, Mn);
        bn_apply<<<(Mn * DIM + 255) / 256, 256, 0, stream>>>(pooled, mu, var, Mn);

        if (layer < 2) {
            hash_clear<<<HASH_CAP / 256, 256, 0, stream>>>(tab, HASH_CAP);
            int* ns = (msk == mskA) ? srcB : srcA;
            int* nd = (msk == mskA) ? dstB : dstA;
            int* nm = (msk == mskA) ? mskB : mskA;
            remap_dedupe<<<(E + 255) / 256, 256, 0, stream>>>(src, dst, msk, ns, nd, nm,
                                                              tab, (unsigned)Mn, E);
            src = ns; dst = nd; msk = nm;
        }
        x = pooled; M = Mn; K = DIM;
    }
}
